// ModConv_69320772158389
// MI455X (gfx1250) — matrix operand order repaired and checked by reading, not compiled, not run on hardware
//
#include <hip/hip_runtime.h>
#include <hip/hip_bf16.h>

// ---------------------------------------------------------------------------
// ModConv (StyleGAN2 modulated conv) for gfx1250 / MI455X
//   style = w @ affine_w^T + affine_b
//   d[b,o] = rsqrt( sum_i style[b,i]^2 * wsq[o,i] + eps )
//   out[b,o] = d[b,o] * conv3x3( style[b,i]*x[b,i], weight )
// Conv decomposed into 9 dense GEMMs (one per tap):
//   out[o,p] += sum_t  W[t][o][:] . xsT[p + shift(t)][:]
// bf16 WMMA (f32 acc). A tiles delivered by the Tensor Data Mover (TDM) with
// hardware LDS padding matching the bank-conflict-free stride; B tiles are
// vector loads with per-pixel boundary zeroing. Double-buffered LDS, one
// barrier per K-step.
// ---------------------------------------------------------------------------

typedef __attribute__((ext_vector_type(8)))  __bf16 v8bf;
typedef __attribute__((ext_vector_type(16))) __bf16 v16bf;
typedef __attribute__((ext_vector_type(8)))  float  v8f;
typedef __attribute__((ext_vector_type(4)))  unsigned int v4u;
typedef __attribute__((ext_vector_type(8)))  int v8i;
typedef __attribute__((ext_vector_type(4)))  int v4i;

#define C_IN   512
#define C_OUT  512
#define HW     1024          // 32*32
#define NKIDX  144           // 9 taps * (512/32) K-steps
#define LDSS   40            // LDS row stride in bf16 = 20 dwords = 80B

// ---------------------------- prep kernels ---------------------------------

__global__ void style_kernel(const float* __restrict__ w,
                             const float* __restrict__ aw,
                             const float* __restrict__ ab,
                             float* __restrict__ style,
                             float* __restrict__ style2) {
  int id = blockIdx.x * blockDim.x + threadIdx.x;   // 8192 = b*512+i
  int b = id >> 9, i = id & 511;
  const float4* wr = reinterpret_cast<const float4*>(w + b * 512);
  const float4* ar = reinterpret_cast<const float4*>(aw + (size_t)i * 512);
  float s = ab[i];
  #pragma unroll 4
  for (int j = 0; j < 128; ++j) {
    float4 a = wr[j], c = ar[j];
    s += a.x * c.x + a.y * c.y + a.z * c.z + a.w * c.w;
  }
  style[id]  = s;
  style2[id] = s * s;
}

__global__ void wsq_kernel(const float* __restrict__ weight,
                           float* __restrict__ wsq) {
  int id = blockIdx.x * blockDim.x + threadIdx.x;   // 262144 = o*512+i
  const float* p = weight + (size_t)id * 9;
  float s = 0.f;
  #pragma unroll
  for (int t = 0; t < 9; ++t) s += p[t] * p[t];
  wsq[id] = s;
}

__global__ void demod_kernel(const float* __restrict__ style2,
                             const float* __restrict__ wsq,
                             float* __restrict__ dmod) {
  int id = blockIdx.x * blockDim.x + threadIdx.x;   // 8192 = b*512+o
  int b = id >> 9, o = id & 511;
  const float4* s2 = reinterpret_cast<const float4*>(style2 + b * 512);
  const float4* wq = reinterpret_cast<const float4*>(wsq + (size_t)o * 512);
  float acc = 1e-8f;
  #pragma unroll 4
  for (int j = 0; j < 128; ++j) {
    float4 a = s2[j], c = wq[j];
    acc += a.x * c.x + a.y * c.y + a.z * c.z + a.w * c.w;
  }
  dmod[id] = rsqrtf(acc);
}

// xsT[b][p][i] = bf16( style[b,i] * x[b,i,p] )   -- LDS-tiled 32x32 transpose
__global__ void xst_kernel(const float* __restrict__ x,
                           const float* __restrict__ style,
                           __bf16* __restrict__ xsT) {
  __shared__ float tile[32][33];
  const int p0 = blockIdx.x * 32;
  const int i0 = blockIdx.y * 32;
  const int b  = blockIdx.z;
  const int tx = threadIdx.x;        // 0..31
  const int ty = threadIdx.y;        // 0..7
  #pragma unroll
  for (int r = 0; r < 4; ++r) {
    int i = i0 + ty + r * 8;
    tile[ty + r * 8][tx] = x[((size_t)b * C_IN + i) * HW + p0 + tx] * style[b * C_IN + i];
  }
  __syncthreads();
  #pragma unroll
  for (int r = 0; r < 4; ++r) {
    int p = p0 + ty + r * 8;
    xsT[((size_t)b * HW + p) * C_IN + i0 + tx] = (__bf16)tile[tx][ty + r * 8];
  }
}

// wre[t][o][i] = bf16( weight[o][i][t] )   -- tap-major reorder
__global__ void wre_kernel(const float* __restrict__ weight,
                           __bf16* __restrict__ wre) {
  int id = blockIdx.x * blockDim.x + threadIdx.x;   // 2359296 = ((t*512)+o)*512+i
  int i  = id & 511;
  int to = id >> 9;
  int o  = to & 511;
  int t  = to >> 9;
  wre[id] = (__bf16)weight[((size_t)o * C_IN + i) * 9 + t];
}

// ------------------- TDM descriptor for one A tile -------------------------
// 2D tile: 128 rows x 32 bf16, row stride 512 elems; LDS padding 4 dwords
// every 16 dwords -> LDS row stride 20 dwords = LDSS bf16.
__device__ __forceinline__ void tdm_load_A(const __bf16* gptr, unsigned lds_off) {
  unsigned long long ga = (unsigned long long)(uintptr_t)gptr;
  v4u g0;
  g0.x = 1u;                                        // count=1 (valid descriptor)
  g0.y = lds_off;                                   // lds_addr (bytes)
  g0.z = (unsigned)ga;                              // global_addr[31:0]
  g0.w = ((unsigned)(ga >> 32) & 0x01FFFFFFu)       // global_addr[56:32]
         | (2u << 30);                              // type=2 ("image")
  v8i g1;
  g1[0] = (int)((1u << 16)      // data_size = 1 -> 2 bytes
              | (1u << 20)      // pad_enable
              | (3u << 22)      // pad_interval: 2^(3+1)=16 dwords
              | (3u << 25));    // pad_amount: 3+1 = 4 dwords
  g1[1] = (int)(512u << 16);    // tensor_dim0[15:0]=512 (bits 63:48)
  g1[2] = (int)(512u << 16);    // tensor_dim0 hi=0; tensor_dim1[15:0]=512
  g1[3] = (int)(32u << 16);     // tensor_dim1 hi=0; tile_dim0=32
  g1[4] = 128;                  // tile_dim1=128; tile_dim2=0
  g1[5] = 512;                  // tensor_dim0_stride=512 (elems)
  g1[6] = 0;
  g1[7] = 0;
  v4i z4 = (v4i){0, 0, 0, 0};
#if defined(__clang_major__) && (__clang_major__ >= 23)
  v8i z8 = (v8i){0, 0, 0, 0, 0, 0, 0, 0};
  __builtin_amdgcn_tensor_load_to_lds(g0, g1, z4, z4, z8, 0);
#else
  __builtin_amdgcn_tensor_load_to_lds(g0, g1, z4, z4, 0);
#endif
}

// ---------------------------- main GEMM kernel -----------------------------
// grid = (8 n-tiles, 4 m-tiles, 16 batch), block = 256 (8 wave32)

__global__ __launch_bounds__(256)
void modconv_gemm(const __bf16* __restrict__ wre,   // [9][512][512] bf16 (A)
                  const __bf16* __restrict__ xsT,   // [16][1024][512] bf16 (B)
                  const float*  __restrict__ dmod,  // [16][512]
                  float* __restrict__ out)          // [16][512][1024]
{
  __shared__ __attribute__((aligned(16))) __bf16 Alds[2][128 * LDSS];
  __shared__ __attribute__((aligned(16))) __bf16 Blds[2][128 * LDSS];

  const int tid   = threadIdx.x;
  const int nbase = blockIdx.x * 128;     // pixel tile
  const int obase = blockIdx.y * 128;     // out-channel tile
  const int bb    = blockIdx.z;           // batch

  const int lane = tid & 31;
  const int wave = tid >> 5;
  const int wm   = wave >> 1;             // 0..3  (M sub-tile)
  const int wn   = wave & 1;              // 0..1  (N sub-tile)
  const int l15  = lane & 15;
  const int hi   = lane >> 4;             // 0/1 half-wave

  // B loader roles: 2 threads per pixel, 16 contiguous channels each
  const int brow = tid >> 1;              // pixel within tile
  const int bseg = (tid & 1) << 4;        // 0 / 16
  const int pix  = nbase + brow;
  const int py   = pix >> 5;
  const int px   = pix & 31;
  const __bf16* xb = xsT + (size_t)bb * (HW * C_IN);

  uint4 br0, br1;

  auto load_regs = [&](int kidx) {        // B: shifted pixel, contiguous channels
    const int t  = kidx >> 4;             // tap (uniform)
    const int kb = (kidx & 15) << 5;      // channel base (uniform)
    const int dy = t / 3 - 1;
    const int dx = t - (t / 3) * 3 - 1;
    const int yy = py + dy, xx = px + dx;
    if (((unsigned)yy < 32u) && ((unsigned)xx < 32u)) {
      const uint4* bp = reinterpret_cast<const uint4*>(
          xb + (size_t)(pix + dy * 32 + dx) * C_IN + kb + bseg);
      br0 = bp[0];
      br1 = bp[1];
    } else {
      br0 = uint4{0u, 0u, 0u, 0u};
      br1 = uint4{0u, 0u, 0u, 0u};
    }
  };

  auto store_tiles = [&](int buf) {
    *reinterpret_cast<uint4*>(&Blds[buf][brow * LDSS + bseg])     = br0;
    *reinterpret_cast<uint4*>(&Blds[buf][brow * LDSS + bseg + 8]) = br1;
  };

  auto issue_tdm = [&](int kidx, int buf) {   // A tile via Tensor Data Mover
    const int t  = kidx >> 4;
    const int kb = (kidx & 15) << 5;
    const __bf16* gp = wre + ((size_t)t * C_OUT + obase) * C_IN + kb;
    tdm_load_A(gp, (unsigned)(uintptr_t)&Alds[buf][0]);
  };

  v8f acc[2][4];
  #pragma unroll
  for (int mi = 0; mi < 2; ++mi)
    #pragma unroll
    for (int ni = 0; ni < 4; ++ni)
      acc[mi][ni] = v8f{0.f, 0.f, 0.f, 0.f, 0.f, 0.f, 0.f, 0.f};

  if (wave == 0) issue_tdm(0, 0);
  load_regs(0);
  store_tiles(0);
  for (int kidx = 0; kidx < NKIDX; ++kidx) {
    __builtin_amdgcn_s_wait_tensorcnt(0);    // A tile for 'cur' has landed
    __syncthreads();                         // one barrier per K-step
    const int cur = kidx & 1;
    if (kidx + 1 < NKIDX) {
      if (wave == 0) issue_tdm(kidx + 1, cur ^ 1);   // overlap DMA with math
      load_regs(kidx + 1);                           // overlap B loads with math
    }

    // A fragment: lanes 0-15 -> K {0..7,16..23}; lanes 16-31 -> K {8..15,24..31}
    v16bf af[2];
    #pragma unroll
    for (int mi = 0; mi < 2; ++mi) {
      const __bf16* ap = &Alds[cur][(wm * 32 + mi * 16 + l15) * LDSS + hi * 8];
      v8bf lo  = *reinterpret_cast<const v8bf*>(ap);
      v8bf hi8 = *reinterpret_cast<const v8bf*>(ap + 16);
      af[mi] = __builtin_shufflevector(lo, hi8, 0,1,2,3,4,5,6,7,8,9,10,11,12,13,14,15);
    }
    // B fragment: lane = column; lanes 0-15 -> K {0..7,16..23}; lanes 16-31 -> K {8..15,24..31}
    v16bf bfm[4];
    #pragma unroll
    for (int ni = 0; ni < 4; ++ni) {
      const __bf16* bp = &Blds[cur][(wn * 64 + ni * 16 + l15) * LDSS + hi * 8];
      v8bf lo  = *reinterpret_cast<const v8bf*>(bp);
      v8bf hi8 = *reinterpret_cast<const v8bf*>(bp + 16);
      bfm[ni] = __builtin_shufflevector(lo, hi8, 0,1,2,3,4,5,6,7,8,9,10,11,12,13,14,15);
    }
    #pragma unroll
    for (int mi = 0; mi < 2; ++mi)
      #pragma unroll
      for (int ni = 0; ni < 4; ++ni)
        acc[mi][ni] = __builtin_amdgcn_wmma_f32_16x16x32_bf16(
            false, af[mi], false, bfm[ni], (short)0, acc[mi][ni], false, false);

    if (kidx + 1 < NKIDX) store_tiles(cur ^ 1);  // fill the other buffer
  }

  // epilogue: scale by d[b,o], store f32
  float dv[2][8];
  #pragma unroll
  for (int mi = 0; mi < 2; ++mi)
    #pragma unroll
    for (int j = 0; j < 8; ++j)
      dv[mi][j] = dmod[bb * C_OUT + obase + wm * 32 + mi * 16 + hi * 8 + j];

  float* outb = out + ((size_t)bb * C_OUT + obase) * HW;
  #pragma unroll
  for (int mi = 0; mi < 2; ++mi)
    #pragma unroll
    for (int ni = 0; ni < 4; ++ni) {
      int col = nbase + wn * 64 + ni * 16 + l15;
      #pragma unroll
      for (int j = 0; j < 8; ++j) {
        int row = wm * 32 + mi * 16 + hi * 8 + j;
        outb[(size_t)row * HW + col] = acc[mi][ni][j] * dv[mi][j];
      }
    }
}

// ------------------------------- launcher ----------------------------------

extern "C" void kernel_launch(void* const* d_in, const int* in_sizes, int n_in,
                              void* d_out, int out_size, void* d_ws, size_t ws_size,
                              hipStream_t stream) {
  (void)in_sizes; (void)n_in; (void)out_size; (void)ws_size;
  const float* x   = (const float*)d_in[0];   // 16*512*1024
  const float* w   = (const float*)d_in[1];   // 16*512
  const float* wgt = (const float*)d_in[2];   // 512*512*9
  const float* aw  = (const float*)d_in[3];   // 512*512
  const float* ab  = (const float*)d_in[4];   // 512
  float* out = (float*)d_out;

  char* ws = (char*)d_ws;
  float*  style  = (float*)(ws);                        //   32 KB
  float*  style2 = (float*)(ws + 32768);                //   32 KB
  float*  dmod   = (float*)(ws + 65536);                //   32 KB
  float*  wsq    = (float*)(ws + 98304);                //    1 MB
  __bf16* wre    = (__bf16*)(ws + 1146880);             //  4.5 MB  [9][512][512]
  __bf16* xsT    = (__bf16*)(ws + 5865472);             //   16 MB  [16][1024][512]

  style_kernel<<<32,   256, 0, stream>>>(w, aw, ab, style, style2);
  wsq_kernel  <<<1024, 256, 0, stream>>>(wgt, wsq);
  demod_kernel<<<32,   256, 0, stream>>>(style2, wsq, dmod);

  dim3 tg(32, 16, 16), tb(32, 8);       // pixel tiles, channel tiles, batch
  xst_kernel  <<<tg, tb, 0, stream>>>(x, style, xsT);
  wre_kernel  <<<9216, 256, 0, stream>>>(wgt, wre);

  dim3 grid(8, 4, 16);                  // n-tiles, m-tiles, batch
  modconv_gemm<<<grid, 256, 0, stream>>>(wre, xsT, dmod, out);
}
